// PhiMoESparseMoeBlock_52578989638363
// MI455X (gfx1250) — matrix operand order repaired and checked by reading, not compiled, not run on hardware
//
#include <hip/hip_runtime.h>
#include <hip/hip_bf16.h>

// ---------------------------------------------------------------------------
// Phi MoE sparse block for MI455X (gfx1250, wave32, WMMA).
// T=512 tokens, H=1024, F=4096, E=8, top-2 routing.
// Roofline: 402MB fp32 weights / 23.3TB/s ~= 17.3us floor. Each weight element
// is loaded + f32->bf16 converted exactly ONCE per (expert, 16-token tile):
// block = (expert, token tile) owns all H output columns; 8 waves split F for
// the w1/w3 GEMMs (shared h via LDS) and split output columns for the w2 GEMM.
// ---------------------------------------------------------------------------

typedef __attribute__((ext_vector_type(16))) __bf16 v16bf;
typedef __attribute__((ext_vector_type(8)))  __bf16 v8bf;
typedef __attribute__((ext_vector_type(8)))  float  v8f;
typedef __attribute__((ext_vector_type(4)))  float  f32x4;

constexpr int T = 512;
constexpr int H = 1024;
constexpr int F = 4096;
constexpr int E = 8;
constexpr int XS = 1032;   // LDS x-tile row stride (halves), 16B aligned rows
constexpr int HS = 264;    // LDS h-tile row stride (halves), 16B aligned rows

__device__ __forceinline__ v8f wmma_bf16(v16bf a, v16bf b, v8f c) {
  return __builtin_amdgcn_wmma_f32_16x16x32_bf16(false, a, false, b,
                                                 (short)0, c, false, false);
}

// Load 16 consecutive f32 from global, convert to bf16 fragment half-row.
__device__ __forceinline__ v16bf ld_frag_g(const float* __restrict__ p) {
  v16bf r;
#pragma unroll
  for (int i = 0; i < 4; ++i) {
    f32x4 v = *(const f32x4*)(p + 4 * i);
    r[4 * i + 0] = (__bf16)v[0];
    r[4 * i + 1] = (__bf16)v[1];
    r[4 * i + 2] = (__bf16)v[2];
    r[4 * i + 3] = (__bf16)v[3];
  }
  return r;
}

// A-fragment (16-bit, 16x32): this lane holds 16 consecutive halves at p.
__device__ __forceinline__ v16bf ld_afrag_lds(const __bf16* p) {
  v8bf lo = *(const v8bf*)p;
  v8bf hi = *(const v8bf*)(p + 8);
  return __builtin_shufflevector(lo, hi, 0, 1, 2, 3, 4, 5, 6, 7,
                                 8, 9, 10, 11, 12, 13, 14, 15);
}

__device__ __forceinline__ float silu(float x) {
  return x * (1.0f / (1.0f + __expf(-x)));
}

// ---------------------------------------------------------------------------
__global__ void moe_init_kernel(float* __restrict__ out, int* __restrict__ counts) {
  int idx = blockIdx.x * 256 + threadIdx.x;
  if (idx < T * H) out[idx] = 0.0f;
  if (idx < E) counts[idx] = 0;
}

// ---------------------------------------------------------------------------
// Gate: one wave per token. Logits -> jitter-masked double softmax (phi
// routing) -> append (token, weight) to per-expert lists.
__global__ void moe_gate_kernel(const float* __restrict__ x,
                                const float* __restrict__ gw,
                                int* __restrict__ counts,
                                int* __restrict__ tok_idx,
                                float* __restrict__ tok_w) {
  const int wv = threadIdx.x >> 5;
  const int lane = threadIdx.x & 31;
  const int t = blockIdx.x * 8 + wv;
  if (t >= T) return;

  float s[E];
#pragma unroll
  for (int e = 0; e < E; ++e) s[e] = 0.0f;

  const float* xp = x + (size_t)t * H;
  for (int k = lane; k < H; k += 32) {
    float xv = xp[k];
#pragma unroll
    for (int e = 0; e < E; ++e) s[e] += xv * gw[e * H + k];
  }
#pragma unroll
  for (int e = 0; e < E; ++e) {
#pragma unroll
    for (int off = 16; off > 0; off >>= 1) s[e] += __shfl_xor(s[e], off, 32);
  }

  if (lane == 0) {
    // top-2 (ties -> lowest index, matching lax.top_k)
    float m1 = -3.0e38f; int s1 = 0;
#pragma unroll
    for (int e = 0; e < E; ++e) if (s[e] > m1) { m1 = s[e]; s1 = e; }
    float m2 = -3.0e38f; int s2 = 0;
#pragma unroll
    for (int e = 0; e < E; ++e) if (e != s1 && s[e] > m2) { m2 = s[e]; s2 = e; }

    const float thr = 0.02f;  // 2 * jitter_eps
    float sum1 = 0.0f;
#pragma unroll
    for (int e = 0; e < E; ++e) {
      float ratio = (m1 - s[e]) / fmaxf(fabsf(s[e]), m1);
      if (!(ratio > thr)) sum1 += __expf(s[e] - m1);
    }
    float mult1 = 1.0f / sum1;

    float sum2 = 0.0f;
#pragma unroll
    for (int e = 0; e < E; ++e) {
      if (e == s1) continue;
      float ratio = (m2 - s[e]) / fmaxf(fabsf(s[e]), m2);
      if (!(ratio > thr)) sum2 += __expf(s[e] - m2);
    }
    float mult2 = 1.0f / sum2;

    int p = atomicAdd(&counts[s1], 1);
    tok_idx[s1 * T + p] = t;
    tok_w[s1 * T + p] = mult1;
    p = atomicAdd(&counts[s2], 1);
    tok_idx[s2 * T + p] = t;
    tok_w[s2 * T + p] = mult2;
  }
}

// ---------------------------------------------------------------------------
// Expert FFN: block = (expert e, 16-token tile), owns all H=1024 out columns.
// Per F-round of 256:
//   phase A: wave w computes h[16 x 32] for f-chunk (round*256 + w*32):
//            a = x@w1^T, b = x@w3^T (K=1024, 32 WMMAs each x2 subtiles),
//            h = silu(a)*b -> bf16 into shared lds_h[16 x 256].
//   phase B: wave w accumulates out cols [w*128, w*128+128) against w2 using
//            all 8 h chunks (64 WMMAs).
// End: routing-weighted global_atomic_add_f32 into out (token has 2 experts).
__global__ void __launch_bounds__(256)
moe_expert_kernel(const float* __restrict__ x,
                  const float* __restrict__ w1,
                  const float* __restrict__ w2,
                  const float* __restrict__ w3,
                  const int* __restrict__ counts,
                  const int* __restrict__ tok_idx,
                  const float* __restrict__ tok_w,
                  float* __restrict__ out) {
  const int e = blockIdx.y;
  const int tile = blockIdx.x;
  const int cnt = counts[e];
  if (tile * 16 >= cnt) return;

  __shared__ __bf16 lds_x[16 * XS];   // x tile, bf16
  __shared__ __bf16 lds_h[16 * HS];   // shared h round buffer (16 x 256 used)
  __shared__ int    s_tok[16];
  __shared__ float  s_w[16];

  const int tid = threadIdx.x;
  const int wv = tid >> 5;
  const int lane = tid & 31;
  const int lm = lane & 15;
  const int hi = lane >> 4;

  if (tid < 16) {
    int g = tile * 16 + tid;
    s_tok[tid] = (g < cnt) ? tok_idx[e * T + g] : -1;
    s_w[tid] = (g < cnt) ? tok_w[e * T + g] : 0.0f;
  }
  __syncthreads();

  // Stage x tile to LDS as bf16 (invalid rows -> 0).
#pragma unroll
  for (int i = 0; i < 64; ++i) {
    int idx = tid + i * 256;
    int r = idx >> 10, c = idx & 1023;
    int tk = s_tok[r];
    float v = (tk >= 0) ? x[(size_t)tk * H + c] : 0.0f;
    lds_x[r * XS + c] = (__bf16)v;
  }
  __syncthreads();

  const float* w1e = w1 + (size_t)e * F * H;
  const float* w3e = w3 + (size_t)e * F * H;
  const float* w2e = w2 + (size_t)e * H * F;

  const v8f vzero = {0.f, 0.f, 0.f, 0.f, 0.f, 0.f, 0.f, 0.f};
  v8f acc[8];
#pragma unroll
  for (int cg = 0; cg < 8; ++cg) acc[cg] = vzero;

  const __bf16* xrow = &lds_x[lm * XS + hi * 16];

  for (int iter = 0; iter < 16; ++iter) {
    const int f0 = iter * 256 + wv * 32;  // this wave's f-chunk this round
    v8f a0 = vzero, a1 = vzero, b0 = vzero, b1 = vzero;

    // B-fragment bases: lane lm -> FFN neuron row (B lane = N), hi -> K half.
    const float* p10 = w1e + (size_t)(f0 + lm) * H + hi * 16;
    const float* p11 = w1e + (size_t)(f0 + 16 + lm) * H + hi * 16;
    const float* p30 = w3e + (size_t)(f0 + lm) * H + hi * 16;
    const float* p31 = w3e + (size_t)(f0 + 16 + lm) * H + hi * 16;

    // Prefetch next round's rows of w1/w3 (global_prefetch_b8 path).
    if (iter + 1 < 16) {
      __builtin_prefetch(p10 + 256 * H, 0, 0);
      __builtin_prefetch(p30 + 256 * H, 0, 0);
    }

#pragma unroll 2
    for (int k0 = 0; k0 < H; k0 += 32) {
      v16bf A = ld_afrag_lds(xrow + k0);          // shared by all 4 GEMMs
      a0 = wmma_bf16(A, ld_frag_g(p10 + k0), a0);
      a1 = wmma_bf16(A, ld_frag_g(p11 + k0), a1);
      b0 = wmma_bf16(A, ld_frag_g(p30 + k0), b0);
      b1 = wmma_bf16(A, ld_frag_g(p31 + k0), b1);
    }

    // h = silu(a)*b -> bf16 into shared lds_h at this wave's 32 columns.
    {
      __bf16* hb = &lds_h[wv * 32];
#pragma unroll
      for (int i = 0; i < 8; ++i) {
        int row = i + hi * 8;
        hb[row * HS + lm]      = (__bf16)(silu(a0[i]) * b0[i]);
        hb[row * HS + 16 + lm] = (__bf16)(silu(a1[i]) * b1[i]);
      }
    }
    __syncthreads();  // all 256 h columns of this round visible

    // Second GEMM: this wave's 128 out cols vs all 8 h chunks of this round.
#pragma unroll
    for (int sub = 0; sub < 8; ++sub) {
      v16bf hA = ld_afrag_lds(&lds_h[lm * HS + sub * 32 + hi * 16]);
      const int kb = iter * 256 + sub * 32 + hi * 16;
#pragma unroll
      for (int cg = 0; cg < 8; ++cg) {
        const float* pw2 = w2e + (size_t)(wv * 128 + cg * 16 + lm) * F + kb;
        acc[cg] = wmma_bf16(hA, ld_frag_g(pw2), acc[cg]);
      }
    }
    __syncthreads();  // before next round overwrites lds_h
  }

  // Routing-weighted accumulation into out (each token hits 2 experts).
#pragma unroll
  for (int cg = 0; cg < 8; ++cg) {
#pragma unroll
    for (int i = 0; i < 8; ++i) {
      int row = i + hi * 8;
      int tk = s_tok[row];
      if (tk >= 0) {
        int col = wv * 128 + cg * 16 + lm;
        atomicAdd(&out[(size_t)tk * H + col], acc[cg][i] * s_w[row]);
      }
    }
  }
}

// ---------------------------------------------------------------------------
extern "C" void kernel_launch(void* const* d_in, const int* in_sizes, int n_in,
                              void* d_out, int out_size, void* d_ws, size_t ws_size,
                              hipStream_t stream) {
  const float* x  = (const float*)d_in[0];   // (1,512,1024)
  const float* gw = (const float*)d_in[1];   // (8,1024)
  const float* w1 = (const float*)d_in[2];   // (8,4096,1024)
  const float* w2 = (const float*)d_in[3];   // (8,1024,4096)
  const float* w3 = (const float*)d_in[4];   // (8,4096,1024)
  float* out = (float*)d_out;                // (1,512,1024) f32

  // Workspace layout
  int*   counts  = (int*)d_ws;                 // [8]
  int*   tok_idx = counts + 8;                 // [8][512]
  float* tok_w   = (float*)(tok_idx + E * T);  // [8][512]

  moe_init_kernel<<<(T * H + 255) / 256, 256, 0, stream>>>(out, counts);
  moe_gate_kernel<<<T / 8, 256, 0, stream>>>(x, gw, counts, tok_idx, tok_w);
  dim3 grid(T / 16, E);  // max token tiles x experts (blocks early-exit)
  moe_expert_kernel<<<grid, 256, 0, stream>>>(x, w1, w2, w3, counts,
                                              tok_idx, tok_w, out);
}
